// Model_65335042507137
// MI455X (gfx1250) — matrix operand order repaired and checked by reading, not compiled, not run on hardware
//
#include <hip/hip_runtime.h>

// ---------------------------------------------------------------------------
// Spec-decode ragged position/verified-id emission for MI455X (gfx1250).
// 3-phase hierarchical scan; phase 2 runs the scan on the WMMA engine
// (v_wmma_f32_16x16x32_f16) with provably exact integer arithmetic:
//   block sums <= 2048  -> exact in f16
//   all prefixes <= 8.4M -> exact in f32 accumulation
// Phases 1/3 are the bandwidth-critical kernels (~55 MB total traffic,
// ~2.4us at 23.3 TB/s) and use wave32 shuffles + LDS only.
// ---------------------------------------------------------------------------

#define BLK 256          // 8 wave32 waves per block
#define WAVES (BLK / 32)

typedef __attribute__((ext_vector_type(16))) _Float16 v16h;
typedef __attribute__((ext_vector_type(8)))  float    v8f;

// ---------------- Phase 1: per-256-element block sums (as float) -----------
__global__ void sd_phase1_block_sums(const int* __restrict__ accept,
                                     float* __restrict__ ws_sums, int B) {
    const int b = blockIdx.x * BLK + threadIdx.x;
    int v = (b < B) ? accept[b] : 0;
#pragma unroll
    for (int d = 16; d >= 1; d >>= 1) v += __shfl_xor(v, d, 32);
    __shared__ int wsum[WAVES];
    const int lane = threadIdx.x & 31;
    const int wave = threadIdx.x >> 5;
    if (lane == 0) wsum[wave] = v;
    __syncthreads();
    if (threadIdx.x == 0) {
        int s = 0;
#pragma unroll
        for (int w = 0; w < WAVES; ++w) s += wsum[w];
        ws_sums[blockIdx.x] = (float)s;   // <= 2048 : exact in f16
    }
}

// ---------------- Phase 2: single-wave WMMA scan of block sums -------------
// D = A(16x32 f16) x B(32x16 f16) + 0, f32 accumulate.
// A: lower-triangular ones in K<16 (constant); B: column j = 16 consecutive
// block sums (chunk j), rows K>=16 zero. D[i][j] = inclusive prefix of chunk j.
__global__ void sd_phase2_wmma_scan(const float* __restrict__ ws_sums,
                                    unsigned* __restrict__ ws_incl, int NB) {
    const int lane = threadIdx.x;     // 0..31, launched with exactly one wave
    const int j    = lane & 15;       // A row index AND B/D column index
    const int hi   = lane >> 4;

    // Constant triangular A in contiguous k order (K = 16*hi + h, same as B):
// lanes 0-15: halves 0-15 -> K=0..15
// lanes16-31: halves 0-15 -> K=16..31
    v16h a;
#pragma unroll
    for (int h = 0; h < 16; ++h) {
        const int K = h + hi * 16;
        a[h] = (K <= j) ? (_Float16)1.0f : (_Float16)0.0f;
    }

    float carry = 0.0f;
    const int iters = (NB + 255) / 256;
    for (int t = 0; t < iters; ++t) {
        const int base = t * 256;

        // B layout (32x16, f16): lanes 0-15 hold K=0..15 of column N=lane,
        // lanes 16-31 hold K=16..31 (we feed zeros there).
        v16h bm;
#pragma unroll
        for (int h = 0; h < 16; ++h) {
            float x = 0.0f;
            if (hi == 0) {
                const int e = base + 16 * j + h;
                if (e < NB) x = ws_sums[e];
            }
            bm[h] = (_Float16)x;
        }

        v8f c = {};
        v8f d = __builtin_amdgcn_wmma_f32_16x16x32_f16(
            /*neg_a=*/false, a, /*neg_b=*/false, bm,
            /*c_mod=*/(short)0, c, /*reuse_a=*/false, /*reuse_b=*/false);

        // Chunk totals T[q] = D[15][q] live in lane 16+q, vgpr 7.
        const float myT = d[7];
        float chunkExcl = 0.0f, blockTotal = 0.0f;
#pragma unroll
        for (int q = 0; q < 16; ++q) {
            const float Tq = __shfl(myT, 16 + q, 32);
            if (j > q) chunkExcl += Tq;
            blockTotal += Tq;
        }

        // D VGPR v holds (M = 8*hi + v, N = j): inclusive prefix of element
        // e = base + 16*j + M. Store global inclusive scan as u32 (exact).
#pragma unroll
        for (int v = 0; v < 8; ++v) {
            const int e = base + 16 * j + 8 * hi + v;
            if (e < NB) ws_incl[e] = (unsigned)(carry + chunkExcl + d[v]);
        }
        carry += blockTotal;
    }
}

// ---------------- Phase 3: local rescan + ragged emission ------------------
__global__ void sd_phase3_emit(const int* __restrict__ verified,
                               const int* __restrict__ seq,
                               const int* __restrict__ accept,
                               const unsigned* __restrict__ ws_incl,
                               float* __restrict__ out,
                               int B, int total) {
    const int tid  = threadIdx.x;
    const int b    = blockIdx.x * BLK + tid;
    const int len  = (b < B) ? accept[b] : 0;
    const int lane = tid & 31;
    const int wave = tid >> 5;

    // wave32 inclusive scan (Hillis-Steele)
    int incl = len;
#pragma unroll
    for (int d = 1; d < 32; d <<= 1) {
        const int t = __shfl_up(incl, d, 32);
        if (lane >= d) incl += t;
    }
    const int laneExcl = incl - len;

    __shared__ int wsum[WAVES];
    if (lane == 31) wsum[wave] = incl;
    __syncthreads();
    int waveExcl = 0;
#pragma unroll
    for (int w = 0; w < WAVES; ++w)
        if (w < wave) waveExcl += wsum[w];

    const unsigned blockOff = (blockIdx.x == 0) ? 0u : ws_incl[blockIdx.x - 1];
    const long start = (long)blockOff + waveExcl + laneExcl;

    if (b < B) {
        const int pos0 = seq[b] - len;          // first emitted position
        for (int i = 0; i < len; ++i)           // adjacent lanes -> adjacent
            out[start + i] = (float)(pos0 + i); // ranges: coalesced stores
        out[(long)total + b] = (float)verified[start + len - 1];
    }
}

// ---------------------------------------------------------------------------
extern "C" void kernel_launch(void* const* d_in, const int* in_sizes, int n_in,
                              void* d_out, int out_size, void* d_ws, size_t ws_size,
                              hipStream_t stream) {
    const int* verified = (const int*)d_in[0];   // [B*MAX_ACCEPT]
    const int* seq      = (const int*)d_in[1];   // [B]
    const int* accept   = (const int*)d_in[2];   // [B]
    const int  B        = in_sizes[1];
    const int  total    = out_size - B;          // ragged positions length
    float*     out      = (float*)d_out;

    const int NB = (B + BLK - 1) / BLK;          // 4096 for B = 2^20

    float*    ws_sums = (float*)d_ws;
    size_t    off     = (((size_t)NB * sizeof(float)) + 255) & ~(size_t)255;
    unsigned* ws_incl = (unsigned*)((char*)d_ws + off);

    sd_phase1_block_sums<<<NB, BLK, 0, stream>>>(accept, ws_sums, B);
    sd_phase2_wmma_scan<<<1, 32, 0, stream>>>(ws_sums, ws_incl, NB);
    sd_phase3_emit<<<NB, BLK, 0, stream>>>(verified, seq, accept, ws_incl,
                                           out, B, total);
}
